// SequentialGraphGenerator_67095979098650
// MI455X (gfx1250) — matrix operand order repaired and checked by reading, not compiled, not run on hardware
//
#include <hip/hip_runtime.h>
#include <hip/hip_bf16.h>
#include <math.h>

// ---------- types ----------
typedef __bf16 bf16_t;
typedef __attribute__((ext_vector_type(16))) __bf16 v16bf;
typedef __attribute__((ext_vector_type(8)))  __bf16 v8bf;
typedef __attribute__((ext_vector_type(8)))  float  v8f;

union BF16Frag { v16bf v; v8bf h[2]; };

#define WMMA_BF16(a, b, c) \
  __builtin_amdgcn_wmma_f32_16x16x32_bf16(false, (a), false, (b), (short)0, (c), false, false)

// Problem constants
#define BB   128
#define NN   512
#define HH   256
#define HS_  64
#define TT   511      // N-1 steps
#define G3H  768      // 3*H

// ---------- fragment loaders (layouts per CDNA5 ISA 7.12.2) ----------
// A (16x32 bf16): lane L: m = L&15, hi = L>>4; element e -> k = (e>>3)*16 + hi*8 + (e&7)
__device__ __forceinline__ v16bf loadAfrag(const bf16_t* A, int lda, int row, int k0, int hi) {
  const bf16_t* p = A + (size_t)row * lda + k0 + hi * 8;
  BF16Frag u;
  u.h[0] = *(const v8bf*)(p);
  u.h[1] = *(const v8bf*)(p + 16);
  return u.v;
}
// B packed: frag (kt, jt): lane holds 16 contiguous bf16 -> one 32B load
__device__ __forceinline__ v16bf loadBfrag(const bf16_t* Bpk, int jtiles, int kt, int jt, int lane) {
  return *(const v16bf*)(Bpk + (((size_t)kt * jtiles + jt) * 32 + (size_t)lane) * 16);
}

__device__ __forceinline__ float softplusf_(float x) {
  return fmaxf(x, 0.f) + log1pf(expf(-fabsf(x)));
}
__device__ __forceinline__ float sigmoidf_(float x) { return 1.f / (1.f + expf(-x)); }

__device__ __forceinline__ v8f zero8() {
  v8f z = {0.f, 0.f, 0.f, 0.f, 0.f, 0.f, 0.f, 0.f};
  return z;
}

// ---------- weight packing: W (J,K) row-major fp32 -> WMMA B-fragment bf16 ----------
__global__ void pack_b_kernel(const float* __restrict__ W, bf16_t* __restrict__ out,
                              int J, int K) {
  int jtiles = J >> 4;
  size_t total = (size_t)J * K;
  for (size_t gid = (size_t)blockIdx.x * blockDim.x + threadIdx.x; gid < total;
       gid += (size_t)gridDim.x * blockDim.x) {
    int e    = (int)(gid & 15);
    int lane = (int)((gid >> 4) & 31);
    size_t tile = gid >> 9;
    int jt = (int)(tile % jtiles);
    int kt = (int)(tile / jtiles);
    int j = jt * 16 + (lane & 15);
    int k = kt * 32 + (e >> 3) * 16 + (lane >> 4) * 8 + (e & 7);
    out[gid] = (bf16_t)W[(size_t)j * K + k];
  }
}

// ---------- GI = X @ W_ih^T + b_ih  (A built on the fly from adjacency) ----------
// grid = (12, 511): rows s*128..+128 (s = blockIdx.y), cols blockIdx.x*64..+64
// K pipeline: double-buffered loads (issue k+1 loads before k's WMMAs).
__global__ __launch_bounds__(256) void gemm_gi_kernel(
    const float* __restrict__ adj, const bf16_t* __restrict__ Wih_pk,
    const float* __restrict__ b_ih, float* __restrict__ GI) {
  const int lane = threadIdx.x & 31, w = threadIdx.x >> 5, hi = lane >> 4;
  const int s = blockIdx.y;
  const int jt0 = blockIdx.x * 4;
  const int jtiles = G3H / 16;  // 48
  const int KT = NN / 32;       // 16 (even)
  v8f acc[4];
#pragma unroll
  for (int j = 0; j < 4; ++j) acc[j] = zero8();

  if (s > 0) {  // s==0 -> x is all zeros (uniform skip; gi = b_ih)
    const int b = w * 16 + (lane & 15);
    const float* xrow = adj + ((size_t)b * NN + s) * NN;  // adj[b, s, :]
    // preload kt = 0
    v8f xa0 = *(const v8f*)(xrow + hi * 8);
    v8f xa1 = *(const v8f*)(xrow + hi * 8 + 16);
    v16bf bb0[4];
#pragma unroll
    for (int j = 0; j < 4; ++j) bb0[j] = loadBfrag(Wih_pk, jtiles, 0, jt0 + j, lane);
#pragma unroll 1
    for (int kt = 0; kt < KT; kt += 2) {
      // issue loads for kt+1
      const float* p1 = xrow + (kt + 1) * 32 + hi * 8;
      v8f xb0 = *(const v8f*)(p1);
      v8f xb1 = *(const v8f*)(p1 + 16);
      v16bf bb1[4];
#pragma unroll
      for (int j = 0; j < 4; ++j) bb1[j] = loadBfrag(Wih_pk, jtiles, kt + 1, jt0 + j, lane);
      // compute kt
      BF16Frag u;
#pragma unroll
      for (int i = 0; i < 8; ++i) { u.h[0][i] = (bf16_t)xa0[i]; u.h[1][i] = (bf16_t)xa1[i]; }
#pragma unroll
      for (int j = 0; j < 4; ++j) acc[j] = WMMA_BF16(u.v, bb0[j], acc[j]);
      // issue loads for kt+2
      if (kt + 2 < KT) {
        const float* p2 = xrow + (kt + 2) * 32 + hi * 8;
        xa0 = *(const v8f*)(p2);
        xa1 = *(const v8f*)(p2 + 16);
#pragma unroll
        for (int j = 0; j < 4; ++j) bb0[j] = loadBfrag(Wih_pk, jtiles, kt + 2, jt0 + j, lane);
      }
      // compute kt+1
      BF16Frag u2;
#pragma unroll
      for (int i = 0; i < 8; ++i) { u2.h[0][i] = (bf16_t)xb0[i]; u2.h[1][i] = (bf16_t)xb1[i]; }
#pragma unroll
      for (int j = 0; j < 4; ++j) acc[j] = WMMA_BF16(u2.v, bb1[j], acc[j]);
    }
  }
  // epilogue
#pragma unroll
  for (int j = 0; j < 4; ++j) {
    int n = blockIdx.x * 64 + j * 16 + (lane & 15);
    float bi = b_ih[n];
#pragma unroll
    for (int v = 0; v < 8; ++v) {
      int row = w * 16 + hi * 8 + v;  // b within step
      GI[((size_t)s * BB + row) * G3H + n] = acc[j][v] + bi;
    }
  }
}

// ---------- persistent sequential GRU scan (1 block, 512 threads = 16 waves) ----------
__global__ __launch_bounds__(512) void gru_scan_kernel(
    const float* __restrict__ GI, const bf16_t* __restrict__ Whh_pk,
    const float* __restrict__ b_hh, const float* __restrict__ h0,
    float* __restrict__ hcur, bf16_t* __restrict__ hbf,
    float* __restrict__ ghs, bf16_t* __restrict__ Hall) {
  const int tid = threadIdx.x, lane = tid & 31, w = tid >> 5, hi = lane >> 4;
  const int jtiles = G3H / 16;        // 48
  const int mstrip = (w & 7) * 16;    // 8 M strips of 16 rows cover B=128
  const int half = w >> 3;            // two waves share each strip, interleave N chunks
  const int arow = mstrip + (lane & 15);
  const int KT = HH / 32;             // 8 (even)

  for (int i = tid; i < BB * HH; i += 512) {
    float v = h0[i & (HH - 1)];
    hcur[i] = v;
    hbf[i] = (bf16_t)v;
  }
  __threadfence_block();
  __syncthreads();

  for (int s = 0; s < TT; ++s) {
    const float* gi = GI + (size_t)s * (BB * G3H);
    // phase 1: gh = h @ W_hh^T + b_hh  -> ghs (L2-resident scratch)
    for (int ch = half; ch < 12; ch += 2) {  // 12 chunks of 64 cols cover 768
      int jt0 = ch * 4;
      v8f acc[4];
#pragma unroll
      for (int j = 0; j < 4; ++j) acc[j] = zero8();
      v16bf a0 = loadAfrag(hbf, HH, arow, 0, hi);
      v16bf bb0[4];
#pragma unroll
      for (int j = 0; j < 4; ++j) bb0[j] = loadBfrag(Whh_pk, jtiles, 0, jt0 + j, lane);
#pragma unroll 1
      for (int kt = 0; kt < KT; kt += 2) {
        v16bf a1 = loadAfrag(hbf, HH, arow, (kt + 1) * 32, hi);
        v16bf bb1[4];
#pragma unroll
        for (int j = 0; j < 4; ++j) bb1[j] = loadBfrag(Whh_pk, jtiles, kt + 1, jt0 + j, lane);
#pragma unroll
        for (int j = 0; j < 4; ++j) acc[j] = WMMA_BF16(a0, bb0[j], acc[j]);
        if (kt + 2 < KT) {
          a0 = loadAfrag(hbf, HH, arow, (kt + 2) * 32, hi);
#pragma unroll
          for (int j = 0; j < 4; ++j) bb0[j] = loadBfrag(Whh_pk, jtiles, kt + 2, jt0 + j, lane);
        }
#pragma unroll
        for (int j = 0; j < 4; ++j) acc[j] = WMMA_BF16(a1, bb1[j], acc[j]);
      }
#pragma unroll
      for (int j = 0; j < 4; ++j) {
        int n = (jt0 + j) * 16 + (lane & 15);
        float bi = b_hh[n];
#pragma unroll
        for (int v = 0; v < 8; ++v) {
          int row = mstrip + hi * 8 + v;
          ghs[(size_t)row * G3H + n] = acc[j][v] + bi;
        }
      }
    }
    __threadfence_block();
    __syncthreads();
    // phase 2: elementwise GRU update (fp32 state)
    for (int i = tid; i < BB * HH; i += 512) {
      int b = i >> 8, c = i & (HH - 1);
      float gir = gi[(size_t)b * G3H + c];
      float giz = gi[(size_t)b * G3H + HH + c];
      float gin = gi[(size_t)b * G3H + 2 * HH + c];
      float ghr = ghs[(size_t)b * G3H + c];
      float ghz = ghs[(size_t)b * G3H + HH + c];
      float ghn = ghs[(size_t)b * G3H + 2 * HH + c];
      float r = sigmoidf_(gir + ghr);
      float z = sigmoidf_(giz + ghz);
      float n = tanhf(gin + r * ghn);
      float hnew = (1.f - z) * n + z * hcur[i];
      hcur[i] = hnew;
      hbf[i] = (bf16_t)hnew;
      Hall[(size_t)s * (BB * HH) + i] = (bf16_t)hnew;
    }
    __threadfence_block();
    __syncthreads();
  }
}

// ---------- generic bf16 GEMM + bias + relu -> bf16 (E1, E2) ----------
// grid = (N/64, 511); A (65408 x K) bf16 row-major; double-buffered K pipeline
__global__ __launch_bounds__(256) void gemm_relu_bf16_kernel(
    const bf16_t* __restrict__ A, const bf16_t* __restrict__ Bpk,
    const float* __restrict__ bias, bf16_t* __restrict__ C,
    int K, int Ncols, int jtiles) {
  const int lane = threadIdx.x & 31, w = threadIdx.x >> 5, hi = lane >> 4;
  const int rowbase = blockIdx.y * 128;
  const int jt0 = blockIdx.x * 4;
  const int arow = rowbase + w * 16 + (lane & 15);
  v8f acc[4];
#pragma unroll
  for (int j = 0; j < 4; ++j) acc[j] = zero8();
  const int ktmax = K >> 5;  // even for all our K
  v16bf a0 = loadAfrag(A, K, arow, 0, hi);
  v16bf bb0[4];
#pragma unroll
  for (int j = 0; j < 4; ++j) bb0[j] = loadBfrag(Bpk, jtiles, 0, jt0 + j, lane);
#pragma unroll 1
  for (int kt = 0; kt < ktmax; kt += 2) {
    v16bf a1 = loadAfrag(A, K, arow, (kt + 1) * 32, hi);
    v16bf bb1[4];
#pragma unroll
    for (int j = 0; j < 4; ++j) bb1[j] = loadBfrag(Bpk, jtiles, kt + 1, jt0 + j, lane);
#pragma unroll
    for (int j = 0; j < 4; ++j) acc[j] = WMMA_BF16(a0, bb0[j], acc[j]);
    if (kt + 2 < ktmax) {
      a0 = loadAfrag(A, K, arow, (kt + 2) * 32, hi);
#pragma unroll
      for (int j = 0; j < 4; ++j) bb0[j] = loadBfrag(Bpk, jtiles, kt + 2, jt0 + j, lane);
    }
#pragma unroll
    for (int j = 0; j < 4; ++j) acc[j] = WMMA_BF16(a1, bb1[j], acc[j]);
  }
#pragma unroll
  for (int j = 0; j < 4; ++j) {
    int n = blockIdx.x * 64 + j * 16 + (lane & 15);
    float bi = bias[n];
#pragma unroll
    for (int v = 0; v < 8; ++v) {
      int row = rowbase + w * 16 + hi * 8 + v;
      C[(size_t)row * Ncols + n] = (bf16_t)fmaxf(acc[j][v] + bi, 0.f);
    }
  }
}

// ---------- edge logits GEMM with fused masked-BCE reduction ----------
// grid = (8, 511); logits = E2 @ We3^T + be3 ; accumulate BCE into acc_edge
__global__ __launch_bounds__(256) void gemm_edge_loss_kernel(
    const bf16_t* __restrict__ E2, const bf16_t* __restrict__ We3_pk,
    const float* __restrict__ be3, const float* __restrict__ adj,
    const int* __restrict__ lengths, float* __restrict__ acc_edge) {
  const int lane = threadIdx.x & 31, w = threadIdx.x >> 5, hi = lane >> 4;
  const int s = blockIdx.y, t = s + 1;
  const int jt0 = blockIdx.x * 4;
  const int jtiles = NN / 16;  // 32
  const bf16_t* Arow = E2 + (size_t)s * BB * HH;
  const int arow = w * 16 + (lane & 15);
  const int KT = HH / 32;  // 8
  v8f acc[4];
#pragma unroll
  for (int j = 0; j < 4; ++j) acc[j] = zero8();
  v16bf a0 = loadAfrag(Arow, HH, arow, 0, hi);
  v16bf bb0[4];
#pragma unroll
  for (int j = 0; j < 4; ++j) bb0[j] = loadBfrag(We3_pk, jtiles, 0, jt0 + j, lane);
#pragma unroll 1
  for (int kt = 0; kt < KT; kt += 2) {
    v16bf a1 = loadAfrag(Arow, HH, arow, (kt + 1) * 32, hi);
    v16bf bb1[4];
#pragma unroll
    for (int j = 0; j < 4; ++j) bb1[j] = loadBfrag(We3_pk, jtiles, kt + 1, jt0 + j, lane);
#pragma unroll
    for (int j = 0; j < 4; ++j) acc[j] = WMMA_BF16(a0, bb0[j], acc[j]);
    if (kt + 2 < KT) {
      a0 = loadAfrag(Arow, HH, arow, (kt + 2) * 32, hi);
#pragma unroll
      for (int j = 0; j < 4; ++j) bb0[j] = loadBfrag(We3_pk, jtiles, kt + 2, jt0 + j, lane);
    }
#pragma unroll
    for (int j = 0; j < 4; ++j) acc[j] = WMMA_BF16(a1, bb1[j], acc[j]);
  }
  float part = 0.f;
#pragma unroll
  for (int j = 0; j < 4; ++j) {
    int n = blockIdx.x * 64 + j * 16 + (lane & 15);
    float bi = be3[n];
    bool colm = n < t;
#pragma unroll
    for (int v = 0; v < 8; ++v) {
      int b = w * 16 + hi * 8 + v;
      float l = acc[j][v] + bi;
      float y = adj[((size_t)b * NN + (s + 1)) * NN + n];  // target
      if (colm && (lengths[b] > t)) part += softplusf_(l) - l * y;
    }
  }
  __shared__ float red[256];
  red[threadIdx.x] = part;
  __syncthreads();
  for (int sft = 128; sft > 0; sft >>= 1) {
    if (threadIdx.x < sft) red[threadIdx.x] += red[threadIdx.x + sft];
    __syncthreads();
  }
  if (threadIdx.x == 0) atomicAdd(acc_edge, red[0]);
}

// ---------- stop head: s1 = relu(H @ Ws1^T + bs1); logit = s1 @ Ws2 + bs2; BCE ----------
// grid = (511); block does full 128x64 tile, then per-row dot + loss
__global__ __launch_bounds__(256) void stop_head_kernel(
    const bf16_t* __restrict__ Hall, const bf16_t* __restrict__ Ws1_pk,
    const float* __restrict__ bs1, const float* __restrict__ Ws2,
    const float* __restrict__ bs2, const int* __restrict__ lengths,
    float* __restrict__ acc_stop) {
  const int lane = threadIdx.x & 31, w = threadIdx.x >> 5, hi = lane >> 4;
  const int s = blockIdx.x, t = s + 1;
  const int jtiles = HS_ / 16;  // 4
  const bf16_t* Arow = Hall + (size_t)s * BB * HH;
  const int arow = w * 16 + (lane & 15);
  const int KT = HH / 32;  // 8
  v8f acc[4];
#pragma unroll
  for (int j = 0; j < 4; ++j) acc[j] = zero8();
  v16bf a0 = loadAfrag(Arow, HH, arow, 0, hi);
  v16bf bb0[4];
#pragma unroll
  for (int j = 0; j < 4; ++j) bb0[j] = loadBfrag(Ws1_pk, jtiles, 0, j, lane);
#pragma unroll 1
  for (int kt = 0; kt < KT; kt += 2) {
    v16bf a1 = loadAfrag(Arow, HH, arow, (kt + 1) * 32, hi);
    v16bf bb1[4];
#pragma unroll
    for (int j = 0; j < 4; ++j) bb1[j] = loadBfrag(Ws1_pk, jtiles, kt + 1, j, lane);
#pragma unroll
    for (int j = 0; j < 4; ++j) acc[j] = WMMA_BF16(a0, bb0[j], acc[j]);
    if (kt + 2 < KT) {
      a0 = loadAfrag(Arow, HH, arow, (kt + 2) * 32, hi);
#pragma unroll
      for (int j = 0; j < 4; ++j) bb0[j] = loadBfrag(Ws1_pk, jtiles, kt + 2, j, lane);
    }
#pragma unroll
    for (int j = 0; j < 4; ++j) acc[j] = WMMA_BF16(a1, bb1[j], acc[j]);
  }
  __shared__ float s1buf[BB * HS_];
#pragma unroll
  for (int j = 0; j < 4; ++j) {
    int n = j * 16 + (lane & 15);
    float bi = bs1[n];
#pragma unroll
    for (int v = 0; v < 8; ++v) {
      int row = w * 16 + hi * 8 + v;
      s1buf[row * HS_ + n] = fmaxf(acc[j][v] + bi, 0.f);
    }
  }
  __syncthreads();
  float part = 0.f;
  if (threadIdx.x < BB) {
    int b = threadIdx.x;
    float d = bs2[0];
#pragma unroll 4
    for (int k = 0; k < HS_; ++k) d += s1buf[b * HS_ + k] * Ws2[k];
    int L = lengths[b];
    if (L >= t + 1) {
      float y = (L == t + 1) ? 1.f : 0.f;
      part = softplusf_(d) - d * y;
    }
  }
  __shared__ float red[256];
  red[threadIdx.x] = part;
  __syncthreads();
  for (int sft = 128; sft > 0; sft >>= 1) {
    if (threadIdx.x < sft) red[threadIdx.x] += red[threadIdx.x + sft];
    __syncthreads();
  }
  if (threadIdx.x == 0) atomicAdd(acc_stop, red[0]);
}

// ---------- misc ----------
__global__ void zero_accs_kernel(float* accs) {
  if (threadIdx.x < 2) accs[threadIdx.x] = 0.f;
}

__global__ __launch_bounds__(512) void finalize_kernel(
    const int* __restrict__ lengths, const float* __restrict__ accs,
    float* __restrict__ out) {
  __shared__ int rede[512];
  __shared__ int reds[512];
  int tid = threadIdx.x;
  int ne = 0, ns = 0;
  if (tid < TT) {
    int t = tid + 1, c1 = 0, c2 = 0;
    for (int b = 0; b < BB; ++b) {
      int L = lengths[b];
      c1 += (L > t) ? 1 : 0;
      c2 += (L >= t + 1) ? 1 : 0;
    }
    ne = t * c1;
    ns = c2;
  }
  rede[tid] = ne;
  reds[tid] = ns;
  __syncthreads();
  for (int sft = 256; sft > 0; sft >>= 1) {
    if (tid < sft) { rede[tid] += rede[tid + sft]; reds[tid] += reds[tid + sft]; }
    __syncthreads();
  }
  if (tid == 0) {
    int nE = rede[0] > 1 ? rede[0] : 1;
    int nS = reds[0] > 1 ? reds[0] : 1;
    float el = accs[0] / (float)nE;
    float sl = accs[1] / (float)nS;
    out[0] = el; out[1] = sl; out[2] = el + sl;
  }
}

// ---------- host launcher ----------
extern "C" void kernel_launch(void* const* d_in, const int* in_sizes, int n_in,
                              void* d_out, int out_size, void* d_ws, size_t ws_size,
                              hipStream_t stream) {
  (void)in_sizes; (void)n_in; (void)out_size; (void)ws_size;
  const float* adj     = (const float*)d_in[0];
  const int*   lengths = (const int*)  d_in[1];
  const float* W_ih    = (const float*)d_in[2];
  const float* W_hh    = (const float*)d_in[3];
  const float* b_ih    = (const float*)d_in[4];
  const float* b_hh    = (const float*)d_in[5];
  const float* We1     = (const float*)d_in[6];
  const float* be1     = (const float*)d_in[7];
  const float* We2     = (const float*)d_in[8];
  const float* be2     = (const float*)d_in[9];
  const float* We3     = (const float*)d_in[10];
  const float* be3     = (const float*)d_in[11];
  const float* Ws1     = (const float*)d_in[12];
  const float* bs1     = (const float*)d_in[13];
  const float* Ws2     = (const float*)d_in[14];
  const float* bs2     = (const float*)d_in[15];
  const float* h0      = (const float*)d_in[16];
  float* out = (float*)d_out;

  char* base = (char*)d_ws;
  size_t off = 0;
  auto take = [&](size_t bytes) -> void* {
    void* p = base + off;
    off = (off + bytes + 255) & ~(size_t)255;
    return p;
  };
  float*  GI      = (float*)  take((size_t)TT * BB * G3H * 4);   // ~201 MB
  bf16_t* Hall    = (bf16_t*) take((size_t)TT * BB * HH * 2);    // ~33 MB
  bf16_t* E1      = (bf16_t*) take((size_t)TT * BB * HH * 2);
  bf16_t* E2      = (bf16_t*) take((size_t)TT * BB * HH * 2);
  bf16_t* Wih_pk  = (bf16_t*) take((size_t)G3H * NN * 2);
  bf16_t* Whh_pk  = (bf16_t*) take((size_t)G3H * HH * 2);
  bf16_t* We1_pk  = (bf16_t*) take((size_t)HH * HH * 2);
  bf16_t* We2_pk  = (bf16_t*) take((size_t)HH * HH * 2);
  bf16_t* We3_pk  = (bf16_t*) take((size_t)NN * HH * 2);
  bf16_t* Ws1_pk  = (bf16_t*) take((size_t)HS_ * HH * 2);
  float*  hcur    = (float*)  take((size_t)BB * HH * 4);
  bf16_t* hbf     = (bf16_t*) take((size_t)BB * HH * 2);
  float*  ghs     = (float*)  take((size_t)BB * G3H * 4);
  float*  accs    = (float*)  take(2 * sizeof(float));

  zero_accs_kernel<<<1, 64, 0, stream>>>(accs);

  // pack weights into WMMA B-fragment layout (bf16)
  auto packGrid = [](size_t elems) {
    size_t blk = (elems + 255) / 256;
    return dim3((unsigned)(blk < 2048 ? blk : 2048));
  };
  pack_b_kernel<<<packGrid((size_t)G3H * NN), 256, 0, stream>>>(W_ih, Wih_pk, G3H, NN);
  pack_b_kernel<<<packGrid((size_t)G3H * HH), 256, 0, stream>>>(W_hh, Whh_pk, G3H, HH);
  pack_b_kernel<<<packGrid((size_t)HH * HH),  256, 0, stream>>>(We1, We1_pk, HH, HH);
  pack_b_kernel<<<packGrid((size_t)HH * HH),  256, 0, stream>>>(We2, We2_pk, HH, HH);
  pack_b_kernel<<<packGrid((size_t)NN * HH),  256, 0, stream>>>(We3, We3_pk, NN, HH);
  pack_b_kernel<<<packGrid((size_t)HS_ * HH), 256, 0, stream>>>(Ws1, Ws1_pk, HS_, HH);

  // 1) time-batched input GEMM: GI[s,b,:] = x_s[b,:] @ W_ih^T + b_ih
  gemm_gi_kernel<<<dim3(G3H / 64, TT), 256, 0, stream>>>(adj, Wih_pk, b_ih, GI);

  // 2) sequential GRU scan (persistent single workgroup, WMMA for gh)
  gru_scan_kernel<<<1, 512, 0, stream>>>(GI, Whh_pk, b_hh, h0, hcur, hbf, ghs, Hall);

  // 3) time-batched edge MLP
  gemm_relu_bf16_kernel<<<dim3(HH / 64, TT), 256, 0, stream>>>(Hall, We1_pk, be1, E1, HH, HH, HH / 16);
  gemm_relu_bf16_kernel<<<dim3(HH / 64, TT), 256, 0, stream>>>(E1, We2_pk, be2, E2, HH, HH, HH / 16);
  gemm_edge_loss_kernel<<<dim3(NN / 64, TT), 256, 0, stream>>>(E2, We3_pk, be3, adj, lengths, accs);

  // 4) stop head (fused GEMM + dot + BCE)
  stop_head_kernel<<<TT, 256, 0, stream>>>(Hall, Ws1_pk, bs1, Ws2, bs2, lengths, accs + 1);

  // 5) normalization counts + final outputs
  finalize_kernel<<<1, 512, 0, stream>>>(lengths, accs, out);
}
